// Transformer_42082089566277
// MI455X (gfx1250) — matrix operand order repaired and checked by reading, not compiled, not run on hardware
//
#include <hip/hip_runtime.h>
#include <hip/hip_bf16.h>
#include <stdint.h>

// Problem dims
#define Bb 4
#define Nn 1024
#define Dd 1024
#define Hh 16
#define DH 64
#define Ff 4096
#define Ll 4
#define Mm (Bb*Nn)   // 4096 rows of activations

typedef __bf16 bf16_t;
typedef __attribute__((ext_vector_type(16))) bf16_t v16bf;
typedef __attribute__((ext_vector_type(8)))  float   v8f;
typedef unsigned short u16;
typedef unsigned int   u32;

union BFrag { v16bf v; u32 u[8]; };

__device__ __forceinline__ u16 f2bf(float f) {
  union { float f; u32 u; } x; x.f = f;
  u32 r = x.u + 0x7FFFu + ((x.u >> 16) & 1u);   // round-to-nearest-even
  return (u16)(r >> 16);
}

__device__ __forceinline__ float gelu_f(float x) {
  const float c = 0.7978845608028654f; // sqrt(2/pi)
  return 0.5f * x * (1.0f + tanhf(c * (x + 0.044715f * x * x * x)));
}

// CDNA5 async DMA: global -> LDS, 16B per lane, tracked by ASYNCcnt.
// GVS addressing: mem = SGPR_base + VGPR_offset ; VDST = per-lane LDS byte addr.
__device__ __forceinline__ void async_ld_b128(u32 lds_addr, u32 goff, const void* gbase) {
  asm volatile("global_load_async_to_lds_b128 %0, %1, %2"
               :: "v"(lds_addr), "v"(goff), "s"(gbase) : "memory");
}
__device__ __forceinline__ void wait_async0() {
  asm volatile("s_wait_asynccnt 0x0" ::: "memory");
}

// ---------------------------------------------------------------------------
// fp32 [K,Nc] -> bf16 [Nc,K] tiled transpose-convert (one-time weight prep)
// blockIdx.z = layer; 256 threads = 32x8
// ---------------------------------------------------------------------------
__global__ void __launch_bounds__(256) transpose_cvt_kernel(const float* __restrict__ src,
                                                            u16* __restrict__ dst,
                                                            int K, int Nc) {
  __shared__ float tile[32][33];
  const size_t lay = (size_t)blockIdx.z * K * Nc;
  const float* s = src + lay;
  u16* d = dst + lay;
  const int bx = blockIdx.x * 32;            // Nc offset
  const int by = blockIdx.y * 32;            // K offset
  const int tx = threadIdx.x & 31, ty = threadIdx.x >> 5;
  #pragma unroll
  for (int j = 0; j < 32; j += 8)
    tile[ty + j][tx] = s[(size_t)(by + ty + j) * Nc + bx + tx];
  __syncthreads();
  #pragma unroll
  for (int j = 0; j < 32; j += 8)
    d[(size_t)(bx + ty + j) * K + by + tx] = f2bf(tile[tx][ty + j]);
}

__global__ void __launch_bounds__(256) copy_f32_kernel(const float* __restrict__ src,
                                                       float* __restrict__ dst, int n) {
  int i = blockIdx.x * 256 + threadIdx.x;
  if (i < n) dst[i] = src[i];
}

// ---------------------------------------------------------------------------
// LayerNorm over D=1024 (one block per row), fp32 in -> bf16 out
// ---------------------------------------------------------------------------
__global__ void __launch_bounds__(256) ln_kernel(const float* __restrict__ x,
                                                 const float* __restrict__ sc,
                                                 const float* __restrict__ sh,
                                                 u16* __restrict__ out) {
  const int row = blockIdx.x;
  const int tid = threadIdx.x;
  const int lane = tid & 31, wv = tid >> 5;
  const float* xr = x + (size_t)row * Dd;
  const int idx = tid * 4;
  float4 d = *reinterpret_cast<const float4*>(xr + idx);

  float s  = d.x + d.y + d.z + d.w;
  float s2 = d.x*d.x + d.y*d.y + d.z*d.z + d.w*d.w;
  #pragma unroll
  for (int o = 16; o > 0; o >>= 1) { s += __shfl_xor(s, o, 32); s2 += __shfl_xor(s2, o, 32); }

  __shared__ float r1[8], r2[8];
  if (lane == 0) { r1[wv] = s; r2[wv] = s2; }
  __syncthreads();
  float ts = 0.f, ts2 = 0.f;
  #pragma unroll
  for (int i = 0; i < 8; ++i) { ts += r1[i]; ts2 += r2[i]; }
  const float mean = ts * (1.0f / Dd);
  const float var  = ts2 * (1.0f / Dd) - mean * mean;
  const float rinv = rsqrtf(var + 1e-5f);

  u16* orow = out + (size_t)row * Dd + idx;
  float4 scv = *reinterpret_cast<const float4*>(sc + idx);
  float4 shv = *reinterpret_cast<const float4*>(sh + idx);
  orow[0] = f2bf(scv.x * (d.x - mean) * rinv + shv.x);
  orow[1] = f2bf(scv.y * (d.y - mean) * rinv + shv.y);
  orow[2] = f2bf(scv.z * (d.z - mean) * rinv + shv.z);
  orow[3] = f2bf(scv.w * (d.w - mean) * rinv + shv.w);
}

// ---------------------------------------------------------------------------
// bf16 GEMM: C[128x128 tile] = A[M,K] * W[K,Nc] + bias, 8 waves/block.
// A row-major bf16; Wt is W pre-transposed to [Nc,K] bf16 so both operand
// tiles stream via global_load_async_to_lds_b128, double-buffered in LDS.
// OUTMODE 0: store bf16   1: GeLU then store bf16   2: fp32 residual add
// ---------------------------------------------------------------------------
template<int OUTMODE>
__global__ void __launch_bounds__(256) gemm_bf16(const u16* __restrict__ A,
                                                 const u16* __restrict__ Wt,
                                                 const float* __restrict__ bias,
                                                 void* __restrict__ out,
                                                 int K, int Nc) {
  // 80B rows: 16B-aligned async chunks, stride 20 words vs 64 banks
  __shared__ u16 As[2][128][40];   // [buf][row][k]
  __shared__ u16 Ws[2][128][40];   // [buf][col][k]
  const int tid  = threadIdx.x;
  const int w    = tid >> 5, lane = tid & 31;
  const int hi   = lane >> 4, ln16 = lane & 15;
  const int rowBase = blockIdx.y * 128;
  const int colBase = blockIdx.x * 128;

  v8f acc[8];
  #pragma unroll
  for (int t = 0; t < 8; ++t)
    #pragma unroll
    for (int r = 0; r < 8; ++r) acc[t][r] = 0.0f;

  // each thread DMAs 2 A chunks + 2 W chunks (16B each) per K-step
  auto stage = [&](int buf, int k0) {
    const u32 aBase = (u32)(uintptr_t)&As[buf][0][0];
    const u32 wBase = (u32)(uintptr_t)&Ws[buf][0][0];
    #pragma unroll
    for (int j = 0; j < 2; ++j) {
      const int id = tid + j * 256;
      const int r  = id >> 2, cc = id & 3;        // 128 rows x 4 chunks
      async_ld_b128(aBase + r * 80 + cc * 16,
                    (u32)(((size_t)(rowBase + r) * K + k0 + cc * 8) * 2), A);
      async_ld_b128(wBase + r * 80 + cc * 16,
                    (u32)(((size_t)(colBase + r) * K + k0 + cc * 8) * 2), Wt);
    }
  };

  stage(0, 0);
  int i = 0;
  for (int k0 = 0; k0 < K; k0 += 32, ++i) {
    wait_async0();          // my async loads for buf (i&1) are complete
    __syncthreads();        // everyone's are -> tile visible; prev compute done
    if (k0 + 32 < K) stage((i + 1) & 1, k0 + 32);   // overlap DMA with WMMA
    const int buf = i & 1;

    BFrag af;
    const int arow = 16 * w + ln16;
    #pragma unroll
    for (int v = 0; v < 8; ++v) {
      const int kk = 2*v + hi*8 + (v >= 4 ? 8 : 0);   // A layout (16-bit, 16x32)
      af.u[v] = *reinterpret_cast<const u32*>(&As[buf][arow][kk]);
    }
    #pragma unroll
    for (int ct = 0; ct < 8; ++ct) {
      BFrag bf;
      const int bcol = 16 * ct + ln16;
      #pragma unroll
      for (int v = 0; v < 8; ++v) {
        const int kk = 2*v + hi*8 + (v >= 4 ? 8 : 0);   // B layout (16-bit, 32x16)
        bf.u[v] = *reinterpret_cast<const u32*>(&Ws[buf][bcol][kk]);
      }
      acc[ct] = __builtin_amdgcn_wmma_f32_16x16x32_bf16(
          false, af.v, false, bf.v, (short)0, acc[ct], false, false);
    }
  }

  #pragma unroll
  for (int ct = 0; ct < 8; ++ct) {
    const int col = colBase + 16 * ct + ln16;
    const float bsv = bias[col];
    #pragma unroll
    for (int r = 0; r < 8; ++r) {
      const int row = rowBase + 16 * w + r + 8 * hi;  // C layout M = r + 8*hi
      float v = acc[ct][r] + bsv;
      if (OUTMODE == 1) v = gelu_f(v);
      if (OUTMODE == 2) {
        float* o = reinterpret_cast<float*>(out);
        o[(size_t)row * Nc + col] += v;
      } else {
        u16* o = reinterpret_cast<u16*>(out);
        o[(size_t)row * Nc + col] = f2bf(v);
      }
    }
  }
}

// ---------------------------------------------------------------------------
// Causal flash attention: grid (N/64, H, B), 128 threads (4 waves),
// each wave owns 16 query rows; key blocks of 32; bf16 WMMA, fp32 softmax.
// K tile streamed via async DMA; V staged transposed; P re-laid out via
// wave-private LDS (in-order per wave). qkv: bf16 [B,N,3D]; o: bf16 [B,N,D].
// ---------------------------------------------------------------------------
__global__ void __launch_bounds__(128) attn_kernel(const u16* __restrict__ qkv,
                                                   u16* __restrict__ o) {
  __shared__ u16 Ks[32][72];      // [key][dh], 144B rows (16B-aligned chunks)
  __shared__ u16 Vs[64][34];      // transposed: [dh][key]
  __shared__ u16 Ps[4][16][34];   // per-wave P tile [q][key32]

  const int b = blockIdx.z, h = blockIdx.y, qt = blockIdx.x;
  const int qbase = qt * 64;
  const int tid = threadIdx.x;
  const int w = tid >> 5, lane = tid & 31;
  const int hi = lane >> 4, ln16 = lane & 15;

  // Load Q fragments once (dh pairs are contiguous in memory)
  BFrag qf[2];
  const int qrow = qbase + 16 * w + ln16;
  const u16* qg = qkv + (size_t)(b * Nn + qrow) * (3 * Dd) + h * DH;
  #pragma unroll
  for (int s = 0; s < 2; ++s)
    #pragma unroll
    for (int v = 0; v < 8; ++v) {
      const int kk = 2*v + hi*8 + (v >= 4 ? 8 : 0);
      qf[s].u[v] = *reinterpret_cast<const u32*>(qg + s * 32 + kk);
    }

  v8f oacc[4];
  float mrun[8], lsum[8];
  #pragma unroll
  for (int t = 0; t < 4; ++t)
    #pragma unroll
    for (int r = 0; r < 8; ++r) oacc[t][r] = 0.0f;
  #pragma unroll
  for (int r = 0; r < 8; ++r) { mrun[r] = -1e30f; lsum[r] = 0.0f; }

  const u32 kBase = (u32)(uintptr_t)&Ks[0][0];
  const int nkb = qbase / 32 + 2;   // causal: only key blocks up to diagonal
  for (int kb = 0; kb < nkb; ++kb) {
    __syncthreads();                 // prior iteration done reading Ks/Vs
    { // stage K via async DMA: 32 keys x 64 dh = 256 x 16B chunks
      #pragma unroll
      for (int j = 0; j < 2; ++j) {
        const int id = tid + j * 128;
        const int kr = id >> 3, cc = id & 7;
        async_ld_b128(kBase + kr * 144 + cc * 16,
                      (u32)(((size_t)(b * Nn + kb * 32 + kr) * (3 * Dd)
                             + Dd + h * DH + cc * 8) * 2), qkv);
      }
    }
    { // stage V transposed (manual: async can't transpose)
      const int kr = tid >> 2;            // 0..31
      const int cs = (tid & 3) * 16;      // 0,16,32,48
      const u16* vg = qkv + (size_t)(b * Nn + kb * 32 + kr) * (3 * Dd)
                      + 2 * Dd + h * DH + cs;
      uint4 e0 = *reinterpret_cast<const uint4*>(vg);
      uint4 e1 = *reinterpret_cast<const uint4*>(vg + 8);
      u16 tmp[16];
      *reinterpret_cast<uint4*>(&tmp[0]) = e0;
      *reinterpret_cast<uint4*>(&tmp[8]) = e1;
      #pragma unroll
      for (int i = 0; i < 16; ++i) Vs[cs + i][kr] = tmp[i];
    }
    wait_async0();
    __syncthreads();

    // S = Q K^T for 32 keys (two 16-key column tiles, two dh steps)
    v8f sa[2];
    #pragma unroll
    for (int kt = 0; kt < 2; ++kt)
      #pragma unroll
      for (int r = 0; r < 8; ++r) sa[kt][r] = 0.0f;
    #pragma unroll
    for (int s = 0; s < 2; ++s) {
      #pragma unroll
      for (int kt = 0; kt < 2; ++kt) {
        BFrag bf;
        const int keyc = kt * 16 + ln16;
        #pragma unroll
        for (int v = 0; v < 8; ++v) {
          const int dh = s * 32 + 2 * v + hi * 8 + (v >= 4 ? 8 : 0);
          bf.u[v] = *reinterpret_cast<const u32*>(&Ks[keyc][dh]);
        }
        sa[kt] = __builtin_amdgcn_wmma_f32_16x16x32_bf16(
            false, qf[s].v, false, bf.v, (short)0, sa[kt], false, false);
      }
    }

    // online softmax update
    const float scale = 0.125f;           // 1/sqrt(64)
    const int key0 = kb * 32 + ln16;
    const int key1 = key0 + 16;
    #pragma unroll
    for (int r = 0; r < 8; ++r) {
      const int qi = qbase + 16 * w + r + 8 * hi;
      float s0 = (key0 <= qi) ? sa[0][r] * scale : -1e10f;
      float s1 = (key1 <= qi) ? sa[1][r] * scale : -1e10f;
      float rm = fmaxf(s0, s1);
      #pragma unroll
      for (int off = 8; off >= 1; off >>= 1) rm = fmaxf(rm, __shfl_xor(rm, off, 32));
      const float mn = fmaxf(mrun[r], rm);
      const float corr = __expf(mrun[r] - mn);
      mrun[r] = mn;
      const float p0 = __expf(s0 - mn);
      const float p1 = __expf(s1 - mn);
      float rs = p0 + p1;
      #pragma unroll
      for (int off = 8; off >= 1; off >>= 1) rs += __shfl_xor(rs, off, 32);
      lsum[r] = lsum[r] * corr + rs;
      #pragma unroll
      for (int t = 0; t < 4; ++t) oacc[t][r] *= corr;
      Ps[w][r + 8 * hi][ln16]      = f2bf(p0);
      Ps[w][r + 8 * hi][16 + ln16] = f2bf(p1);
    }

    // O += P * V (wave-private LDS is in-order: writes above precede reads)
    BFrag pf;
    #pragma unroll
    for (int v = 0; v < 8; ++v) {
      const int kk = 2*v + hi*8 + (v >= 4 ? 8 : 0);
      pf.u[v] = *reinterpret_cast<const u32*>(&Ps[w][ln16][kk]);
    }
    #pragma unroll
    for (int t = 0; t < 4; ++t) {
      BFrag vf;
      #pragma unroll
      for (int v = 0; v < 8; ++v) {
        const int key = 2 * v + hi * 8 + (v >= 4 ? 8 : 0);
        vf.u[v] = *reinterpret_cast<const u32*>(&Vs[t * 16 + ln16][key]);
      }
      oacc[t] = __builtin_amdgcn_wmma_f32_16x16x32_bf16(
          false, pf.v, false, vf.v, (short)0, oacc[t], false, false);
    }
  }

  // write O (bf16), heads re-interleaved as [B,N,D]
  #pragma unroll
  for (int t = 0; t < 4; ++t)
    #pragma unroll
    for (int r = 0; r < 8; ++r) {
      const int qi = qbase + 16 * w + r + 8 * hi;
      const float val = oacc[t][r] / lsum[r];
      o[(size_t)(b * Nn + qi) * Dd + h * DH + t * 16 + ln16] = f2bf(val);
    }
}

// ---------------------------------------------------------------------------
// Host-side orchestration
// ---------------------------------------------------------------------------
extern "C" void kernel_launch(void* const* d_in, const int* in_sizes, int n_in,
                              void* d_out, int out_size, void* d_ws, size_t ws_size,
                              hipStream_t stream) {
  (void)in_sizes; (void)n_in; (void)out_size; (void)ws_size;

  const float* in_x     = (const float*)d_in[0];
  // d_in[1] = mask_f : unused (causal mask computed analytically)
  const float* ln1_s    = (const float*)d_in[2];
  const float* ln1_b    = (const float*)d_in[3];
  const float* qkv_w    = (const float*)d_in[4];
  const float* qkv_b    = (const float*)d_in[5];
  const float* out_w    = (const float*)d_in[6];
  const float* out_b    = (const float*)d_in[7];
  const float* ln2_s    = (const float*)d_in[8];
  const float* ln2_b    = (const float*)d_in[9];
  const float* w1       = (const float*)d_in[10];
  const float* b1       = (const float*)d_in[11];
  const float* w2       = (const float*)d_in[12];
  const float* b2       = (const float*)d_in[13];

  // workspace carve-up
  char* p = (char*)d_ws;
  float* x_cur  = (float*)p;                 p += (size_t)Mm * Dd * 4;       // 16 MB
  u16* h_bf     = (u16*)p;                   p += (size_t)Mm * Dd * 2;       //  8 MB
  u16* qkv_bf   = (u16*)p;                   p += (size_t)Mm * 3 * Dd * 2;   // 24 MB
  u16* o_bf     = (u16*)p;                   p += (size_t)Mm * Dd * 2;       //  8 MB
  u16* hid_bf   = (u16*)p;                   p += (size_t)Mm * Ff * 2;       // 32 MB
  u16* wqkv_t   = (u16*)p;                   p += (size_t)Ll * Dd * 3*Dd * 2; // [3D,D] per layer
  u16* wout_t   = (u16*)p;                   p += (size_t)Ll * Dd * Dd * 2;   // [D,D]
  u16* w1_t     = (u16*)p;                   p += (size_t)Ll * Dd * Ff * 2;   // [F,D]
  u16* w2_t     = (u16*)p;                   p += (size_t)Ll * Ff * Dd * 2;   // [D,F]

  // x_cur = input x
  hipMemcpyAsync(x_cur, in_x, (size_t)Mm * Dd * 4, hipMemcpyDeviceToDevice, stream);

  // one-time weight transpose+convert fp32 [K,Nc] -> bf16 [Nc,K]
  transpose_cvt_kernel<<<dim3(3 * Dd / 32, Dd / 32, Ll), 256, 0, stream>>>(qkv_w, wqkv_t, Dd, 3 * Dd);
  transpose_cvt_kernel<<<dim3(Dd / 32, Dd / 32, Ll),     256, 0, stream>>>(out_w, wout_t, Dd, Dd);
  transpose_cvt_kernel<<<dim3(Ff / 32, Dd / 32, Ll),     256, 0, stream>>>(w1, w1_t, Dd, Ff);
  transpose_cvt_kernel<<<dim3(Dd / 32, Ff / 32, Ll),     256, 0, stream>>>(w2, w2_t, Ff, Dd);

  for (int l = 0; l < Ll; ++l) {
    // LN1 -> h_bf (bf16)
    ln_kernel<<<Mm, 256, 0, stream>>>(x_cur, ln1_s + (size_t)l * Dd, ln1_b + (size_t)l * Dd, h_bf);
    // QKV projection: [4096,1024] x [1024,3072] -> bf16
    gemm_bf16<0><<<dim3(3 * Dd / 128, Mm / 128), 256, 0, stream>>>(
        h_bf, wqkv_t + (size_t)l * Dd * 3 * Dd, qkv_b + (size_t)l * 3 * Dd,
        (void*)qkv_bf, Dd, 3 * Dd);
    // causal flash attention -> o_bf (bf16)
    attn_kernel<<<dim3(Nn / 64, Hh, Bb), 128, 0, stream>>>(qkv_bf, o_bf);
    // out projection + residual: x_cur += o @ out_w + out_b
    gemm_bf16<2><<<dim3(Dd / 128, Mm / 128), 256, 0, stream>>>(
        o_bf, wout_t + (size_t)l * Dd * Dd, out_b + (size_t)l * Dd,
        (void*)x_cur, Dd, Dd);
    // LN2 -> h_bf
    ln_kernel<<<Mm, 256, 0, stream>>>(x_cur, ln2_s + (size_t)l * Dd, ln2_b + (size_t)l * Dd, h_bf);
    // MLP1 + GeLU: [4096,1024] x [1024,4096] -> bf16 hidden
    gemm_bf16<1><<<dim3(Ff / 128, Mm / 128), 256, 0, stream>>>(
        h_bf, w1_t + (size_t)l * Dd * Ff, b1 + (size_t)l * Ff,
        (void*)hid_bf, Dd, Ff);
    // MLP2 + residual: x_cur += hidden @ w2 + b2
    gemm_bf16<2><<<dim3(Dd / 128, Mm / 128), 256, 0, stream>>>(
        hid_bf, w2_t + (size_t)l * Ff * Dd, b2 + (size_t)l * Dd,
        (void*)x_cur, Ff, Dd);
  }

  // final output (fp32)
  {
    const int n = Mm * Dd;
    copy_f32_kernel<<<(n + 255) / 256, 256, 0, stream>>>(x_cur, (float*)d_out, n);
  }
}
